// RegionProposalNetwork_16277926052392
// MI455X (gfx1250) — matrix operand order repaired and checked by reading, not compiled, not run on hardware
//
#include <hip/hip_runtime.h>

typedef __attribute__((ext_vector_type(16))) _Float16 v16h;
typedef __attribute__((ext_vector_type(8)))  float    v8f;

#define ANUM 3
#define TK 1000
#define NEG_INF (-__builtin_inff())

// ---------------------------------------------------------------------------
// 0) zero page for halo (async loads cannot zero-fill OOB taps)
// ---------------------------------------------------------------------------
__global__ void zero_page(float* zp) {
  zp[threadIdx.x] = 0.0f;  // 64 threads * 4B = 256B of zeros
}

// ---------------------------------------------------------------------------
// 1) NCHW fp32 -> NHWC fp16 feature conversion
// ---------------------------------------------------------------------------
__global__ void cvt_feat(const float* __restrict__ in, _Float16* __restrict__ out,
                         int HW, int total) {
  int idx = blockIdx.x * 256 + threadIdx.x;
  if (idx >= total) return;
  int c = idx & 255;
  int p = (idx >> 8) % HW;
  int n = (idx >> 8) / HW;
  out[idx] = (_Float16)in[((size_t)(n * 256 + c)) * HW + p];
}

// ---------------------------------------------------------------------------
// 2) conv_w (Cout,Cin,3,3) fp32 -> WpT[tap][cout][cin] fp16
// ---------------------------------------------------------------------------
__global__ void pack_w(const float* __restrict__ w, _Float16* __restrict__ out,
                       int total) {
  int idx = blockIdx.x * 256 + threadIdx.x;
  if (idx >= total) return;
  int cin  = idx & 255;
  int cout = (idx >> 8) & 255;
  int tap  = idx >> 16;
  out[idx] = (_Float16)w[((size_t)(cout * 256 + cin)) * 9 + tap];
}

// ---------------------------------------------------------------------------
// 3) 3x3 conv as 9 shifted GEMMs, WMMA f16, double-buffered LDS fed by
//    global_load_async_to_lds_b128 (ASYNCcnt ping-pong pipeline).
//    Block: 256 threads (8 waves), tile 128(M) x 64(N), K-step 32.
// ---------------------------------------------------------------------------
#define BM 128
#define BN 64
#define LDA 40   // halves; row stride 80B (16B-aligned chunks)
#define LDB 40

#define ASYNC_B128(lds32, gaddr64) \
  asm volatile("global_load_async_to_lds_b128 %0, %1, off" \
               :: "v"(lds32), "v"(gaddr64) : "memory")

__global__ __launch_bounds__(256)
void conv3x3_wmma(const _Float16* __restrict__ feat, const _Float16* __restrict__ wpt,
                  const float* __restrict__ bias, const _Float16* __restrict__ zpage,
                  _Float16* __restrict__ tout, int M, int HW, int H, int W) {
  __shared__ __align__(16) _Float16 sA[2][BM * LDA];
  __shared__ __align__(16) _Float16 sB[2][BN * LDB];

  int tid  = threadIdx.x;
  int lane = tid & 31;
  int wv   = tid >> 5;
  int mw   = wv & 3;       // wave's 32-row block within 128
  int nw   = wv >> 2;      // wave's 32-col block within 64
  int mTile = blockIdx.x * BM;
  int nTile = blockIdx.y * BN;

  // A-tile load assignment: thread -> (row, 16-channel chunk)
  int ar  = tid >> 1;
  int apc = tid & 1;
  int am = mTile + ar;
  int an = am / HW;
  int rem = am - an * HW;
  int ah = rem / W;
  int aw = rem - ah * W;
  bool amv = (am < M);

  // B-tile load assignment: thread -> (cout row, 8-k chunk)
  int br = tid >> 2;
  int bk = (tid & 3) << 3;
  int bco = nTile + br;

  // 32-bit LDS byte addresses for async destinations (low dword of generic ptr)
  unsigned baseA0 = (unsigned)(size_t)(const void*)&sA[0][0];
  unsigned baseA1 = (unsigned)(size_t)(const void*)&sA[1][0];
  unsigned baseB0 = (unsigned)(size_t)(const void*)&sB[0][0];
  unsigned baseB1 = (unsigned)(size_t)(const void*)&sB[1][0];
  unsigned offA = (unsigned)(ar * 80 + apc * 32);
  unsigned offB = (unsigned)(br * 80 + (bk << 1));
  unsigned long long zaddr = (unsigned long long)(const void*)zpage;

  v8f acc[2][2] = {};

  // issue the 3 async copies (2xA halo-safe, 1xB) for pipeline step s into buf
  auto issue = [&](int s, int buf) {
    int tap = s >> 3;
    int kc  = (s & 7) << 5;
    int ky = tap / 3 - 1, kx = tap % 3 - 1;
    int hh = ah + ky, ww = aw + kx;
    bool rv = amv && hh >= 0 && hh < H && ww >= 0 && ww < W;
    const _Float16* asrc =
        rv ? (feat + (((size_t)(an * HW + hh * W + ww)) << 8) + kc + apc * 16) : zpage;
    unsigned long long ga = (unsigned long long)(const void*)asrc;
    unsigned lA = (buf ? baseA1 : baseA0) + offA;
    ASYNC_B128(lA, ga);
    ASYNC_B128(lA + 16u, ga + 16ull);
    const _Float16* bsrc = wpt + (((size_t)(tap * 256 + bco)) << 8) + kc + bk;
    unsigned lB = (buf ? baseB1 : baseB0) + offB;
    ASYNC_B128(lB, (unsigned long long)(const void*)bsrc);
  };

  issue(0, 0);

  for (int s = 0; s < 72; ++s) {
    int buf = s & 1;
    if (s + 1 < 72) {
      issue(s + 1, buf ^ 1);
      asm volatile("s_wait_asynccnt 0x3" ::: "memory");  // older group of 3 done
    } else {
      asm volatile("s_wait_asynccnt 0x0" ::: "memory");
    }
    __syncthreads();

    const _Float16* As = &sA[buf][0];
    const _Float16* Bs = &sB[buf][0];
    union Frag { v16h v; uint4 q[2]; };
    int l15 = lane & 15;
    int klo = (lane < 16) ? 0 : 8;    // A: lanes 0-15 K=0..7/16..23, 16-31 K=8..15/24..31
    int koB = (lane < 16) ? 0 : 8;    // B: lanes 0-15 K=0..7/16..23, 16-31 K=8..15/24..31
    Frag a[2], b[2];
#pragma unroll
    for (int i = 0; i < 2; ++i) {
      int row = mw * 32 + i * 16 + l15;
      a[i].q[0] = *(const uint4*)&As[row * LDA + klo];
      a[i].q[1] = *(const uint4*)&As[row * LDA + 16 + klo];
    }
#pragma unroll
    for (int j = 0; j < 2; ++j) {
      int col = nw * 32 + j * 16 + l15;
      b[j].q[0] = *(const uint4*)&Bs[col * LDB + koB];
      b[j].q[1] = *(const uint4*)&Bs[col * LDB + 16 + koB];
    }
#pragma unroll
    for (int i = 0; i < 2; ++i)
#pragma unroll
      for (int j = 0; j < 2; ++j)
        acc[i][j] = __builtin_amdgcn_wmma_f32_16x16x32_f16(
            false, a[i].v, false, b[j].v, (short)0, acc[i][j], false, false);

    __syncthreads();  // reads done before buf is overwritten at step s+1's issue
  }

  // epilogue: bias + relu -> fp16 NHWC t
  int l15 = lane & 15;
  int rhi = (lane < 16) ? 0 : 8;
#pragma unroll
  for (int i = 0; i < 2; ++i) {
#pragma unroll
    for (int j = 0; j < 2; ++j) {
      int col = nTile + nw * 32 + j * 16 + l15;
      float bv = bias[col];
#pragma unroll
      for (int r = 0; r < 8; ++r) {
        int row = mTile + mw * 32 + i * 16 + rhi + r;
        if (row < M) {
          float v = acc[i][j][r] + bv;
          v = v > 0.f ? v : 0.f;
          tout[(((size_t)row) << 8) + col] = (_Float16)v;
        }
      }
    }
  }
}

// ---------------------------------------------------------------------------
// 4) 1x1 heads: obj (3 ch) + deltas (12 ch). One wave per pixel row.
// ---------------------------------------------------------------------------
__global__ __launch_bounds__(256)
void heads(const _Float16* __restrict__ t, const float* __restrict__ clsw,
           const float* __restrict__ clsb, const float* __restrict__ bw,
           const float* __restrict__ bb, float* __restrict__ obj,
           float* __restrict__ deltas, int M, int HW) {
  int lane = threadIdx.x & 31;
  int wv = threadIdx.x >> 5;
  int m = blockIdx.x * 8 + wv;
  if (m >= M) return;
  const _Float16* trow = t + (((size_t)m) << 8);
  float tv[8];
#pragma unroll
  for (int k = 0; k < 8; ++k) tv[k] = (float)trow[lane * 8 + k];
  int n = m / HW, p = m - n * HW;
  int L = HW * ANUM;
  for (int o = 0; o < 15; ++o) {
    const float* wr = (o < 3) ? (clsw + (size_t)o * 256) : (bw + (size_t)(o - 3) * 256);
    float s = 0.f;
#pragma unroll
    for (int k = 0; k < 8; ++k) s += tv[k] * wr[lane * 8 + k];
    for (int off = 16; off > 0; off >>= 1) s += __shfl_xor(s, off, 32);
    if (lane == 0) {
      if (o < 3) {
        obj[(size_t)n * L + p * ANUM + o] = s + clsb[o];
      } else {
        int od = o - 3, a = od >> 2, c = od & 3;
        deltas[(((size_t)n * L + p * ANUM + a) << 2) + c] = s + bb[od];
      }
    }
  }
}

// ---------------------------------------------------------------------------
// 5) per (level,image): top-1000 via threshold bisection + bitonic sort,
//    decode + clip + sigmoid + min-size filter, resort by score.
// ---------------------------------------------------------------------------
__global__ __launch_bounds__(1024)
void topk_level(const float* __restrict__ obj, const float* __restrict__ deltas,
                const float* __restrict__ anchors, int L,
                float* __restrict__ outBoxes, float* __restrict__ outScores) {
  __shared__ float sred[1024];
  __shared__ int sredi[1024];
  __shared__ int scnt;
  __shared__ float cscore[2048];
  __shared__ int cidx[2048];
  __shared__ float4 sbox[TK];
  __shared__ float skey[1024];
  __shared__ int spos[1024];

  int tid = threadIdx.x;
  int n = blockIdx.x;
  const float* o = obj + (size_t)n * L;

  float vmax = -1e30f, vmin = 1e30f;
  for (int i = tid; i < L; i += 1024) {
    float v = o[i];
    vmax = fmaxf(vmax, v);
    vmin = fminf(vmin, v);
  }
  sred[tid] = vmax; __syncthreads();
  for (int s = 512; s > 0; s >>= 1) { if (tid < s) sred[tid] = fmaxf(sred[tid], sred[tid + s]); __syncthreads(); }
  vmax = sred[0]; __syncthreads();
  sred[tid] = vmin; __syncthreads();
  for (int s = 512; s > 0; s >>= 1) { if (tid < s) sred[tid] = fminf(sred[tid], sred[tid + s]); __syncthreads(); }
  vmin = sred[0]; __syncthreads();

  float lo = vmin - 1.0f, hi = vmax;
  for (int it = 0; it < 30; ++it) {
    float mid = 0.5f * (lo + hi);
    int c = 0;
    for (int i = tid; i < L; i += 1024) c += (o[i] > mid) ? 1 : 0;
    sredi[tid] = c; __syncthreads();
    for (int s = 512; s > 0; s >>= 1) { if (tid < s) sredi[tid] += sredi[tid + s]; __syncthreads(); }
    int cnt = sredi[0]; __syncthreads();
    if (cnt >= TK) lo = mid; else hi = mid;
  }
  float thr = lo;

  if (tid == 0) scnt = 0;
  __syncthreads();
  for (int i = tid; i < L; i += 1024) {
    float v = o[i];
    if (v > thr) {
      int p = atomicAdd(&scnt, 1);
      if (p < 2048) { cscore[p] = v; cidx[p] = i; }
    }
  }
  __syncthreads();
  int cnt = scnt < 2048 ? scnt : 2048;
  for (int i = cnt + tid; i < 2048; i += 1024) { cscore[i] = NEG_INF; cidx[i] = 0; }
  __syncthreads();

  // bitonic sort 2048 descending by score
  for (int k = 2; k <= 2048; k <<= 1)
    for (int j = k >> 1; j > 0; j >>= 1) {
      for (int i = tid; i < 2048; i += 1024) {
        int ixj = i ^ j;
        if (ixj > i) {
          float a = cscore[i], b = cscore[ixj];
          bool sw = ((i & k) == 0) ? (a < b) : (a > b);
          if (sw) {
            cscore[i] = b; cscore[ixj] = a;
            int t = cidx[i]; cidx[i] = cidx[ixj]; cidx[ixj] = t;
          }
        }
      }
      __syncthreads();
    }

  // decode top 1000
  if (tid < TK) {
    int si = cidx[tid];
    float logit = cscore[tid];
    float4 a4 = ((const float4*)anchors)[si];
    const float* dd = deltas + (((size_t)n * L + si) << 2);
    float dx = dd[0], dy = dd[1], dw = dd[2], dh = dd[3];
    float wa = a4.z - a4.x, ha = a4.w - a4.y;
    float cxa = a4.x + 0.5f * wa, cya = a4.y + 0.5f * ha;
    const float XCLIP = 4.135166556742356f;
    dw = fminf(dw, XCLIP); dh = fminf(dh, XCLIP);
    float cx = dx * wa + cxa, cy = dy * ha + cya;
    float w2 = __expf(dw) * wa, h2 = __expf(dh) * ha;
    float x0 = fminf(fmaxf(cx - 0.5f * w2, 0.f), 800.f);
    float y0 = fminf(fmaxf(cy - 0.5f * h2, 0.f), 800.f);
    float x1 = fminf(fmaxf(cx + 0.5f * w2, 0.f), 800.f);
    float y1 = fminf(fmaxf(cy + 0.5f * h2, 0.f), 800.f);
    float sc = 1.f / (1.f + __expf(-logit));
    bool valid = ((x1 - x0) >= 1.f) && ((y1 - y0) >= 1.f) && (sc > 0.f);
    sc = valid ? sc : NEG_INF;
    sbox[tid] = make_float4(x0, y0, x1, y1);
    skey[tid] = sc; spos[tid] = tid;
  } else {
    skey[tid] = NEG_INF; spos[tid] = tid;
  }
  __syncthreads();

  // re-sort 1024 by filtered score (argsort(-scores) reorder)
  for (int k = 2; k <= 1024; k <<= 1)
    for (int j = k >> 1; j > 0; j >>= 1) {
      int i = tid, ixj = i ^ j;
      if (ixj > i) {
        float a = skey[i], b = skey[ixj];
        bool sw = ((i & k) == 0) ? (a < b) : (a > b);
        if (sw) {
          skey[i] = b; skey[ixj] = a;
          int t = spos[i]; spos[i] = spos[ixj]; spos[ixj] = t;
        }
      }
      __syncthreads();
    }

  if (tid < TK) {
    int p = spos[tid];
    float4 bx = (p < TK) ? sbox[p] : make_float4(0.f, 0.f, 0.f, 0.f);
    ((float4*)outBoxes)[(size_t)n * TK + tid] = bx;
    outScores[(size_t)n * TK + tid] = skey[tid];
  }
}

// ---------------------------------------------------------------------------
// 6) NMS per (level,image) slot. Serial i, parallel j.
// ---------------------------------------------------------------------------
__global__ __launch_bounds__(1024)
void nms_kernel(const float* __restrict__ boxes, float* __restrict__ scores) {
  __shared__ float4 sb[TK];
  __shared__ float sa[TK];
  __shared__ int keep[TK];
  int tid = threadIdx.x;
  size_t base = (size_t)blockIdx.x * TK;
  if (tid < TK) {
    float4 b = ((const float4*)boxes)[base + tid];
    sb[tid] = b;
    sa[tid] = (b.z - b.x) * (b.w - b.y);
    keep[tid] = 1;
  }
  __syncthreads();
  for (int i = 0; i < TK; ++i) {
    if (keep[i]) {
      float4 bi = sb[i];
      float ai = sa[i];
      for (int j = i + 1 + tid; j < TK; j += 1024) {
        float4 bj = sb[j];
        float xx0 = fmaxf(bi.x, bj.x), yy0 = fmaxf(bi.y, bj.y);
        float xx1 = fminf(bi.z, bj.z), yy1 = fminf(bi.w, bj.w);
        float iw = fmaxf(xx1 - xx0, 0.f), ih = fmaxf(yy1 - yy0, 0.f);
        float inter = iw * ih;
        float iou = inter / (ai + sa[j] - inter + 1e-9f);
        if (iou > 0.7f) keep[j] = 0;
      }
    }
    __syncthreads();
  }
  if (tid < TK && !keep[tid]) scores[base + tid] = NEG_INF;
}

// ---------------------------------------------------------------------------
// 7) final per-image top-1000 across 3 levels (3000 -> pad 4096, bitonic)
// ---------------------------------------------------------------------------
__global__ __launch_bounds__(1024)
void final_topk(const float* __restrict__ lvlBoxes, const float* __restrict__ lvlScores,
                float* __restrict__ out) {
  __shared__ float key[4096];
  __shared__ int kid[4096];
  int tid = threadIdx.x;
  int n = blockIdx.x;
  for (int i = tid; i < 4096; i += 1024) {
    float s = NEG_INF;
    int id = 0;
    if (i < 3 * TK) {
      int li = i / TK, t = i - li * TK;
      int slot = li * 4 + n;
      s = lvlScores[(size_t)slot * TK + t];
      id = slot * TK + t;
    }
    key[i] = s; kid[i] = id;
  }
  __syncthreads();
  for (int k = 2; k <= 4096; k <<= 1)
    for (int j = k >> 1; j > 0; j >>= 1) {
      for (int i = tid; i < 4096; i += 1024) {
        int ixj = i ^ j;
        if (ixj > i) {
          float a = key[i], b = key[ixj];
          bool sw = ((i & k) == 0) ? (a < b) : (a > b);
          if (sw) {
            key[i] = b; key[ixj] = a;
            int t = kid[i]; kid[i] = kid[ixj]; kid[ixj] = t;
          }
        }
      }
      __syncthreads();
    }
  if (tid < TK) {
    int id = kid[tid];
    float4 bx = ((const float4*)lvlBoxes)[id];
    ((float4*)out)[(size_t)n * TK + tid] = bx;            // top_boxes (4,1000,4)
    out[16000 + (size_t)n * TK + tid] = key[tid];         // top_scores (4,1000)
  }
}

// ---------------------------------------------------------------------------
// launch
// ---------------------------------------------------------------------------
extern "C" void kernel_launch(void* const* d_in, const int* in_sizes, int n_in,
                              void* d_out, int out_size, void* d_ws, size_t ws_size,
                              hipStream_t stream) {
  (void)in_sizes; (void)n_in; (void)out_size; (void)ws_size;
  const float* featsF[3]  = {(const float*)d_in[0], (const float*)d_in[2], (const float*)d_in[4]};
  const float* anchors[3] = {(const float*)d_in[1], (const float*)d_in[3], (const float*)d_in[5]};
  const float* conv_w = (const float*)d_in[6];
  const float* conv_b = (const float*)d_in[7];
  const float* cls_w  = (const float*)d_in[8];
  const float* cls_b  = (const float*)d_in[9];
  const float* bbox_w = (const float*)d_in[10];
  const float* bbox_b = (const float*)d_in[11];

  const int Hs[3] = {100, 50, 25};
  const int Ws[3] = {100, 50, 25};
  const int HWs[3] = {10000, 2500, 625};
  const size_t featHOff[3]  = {0, 10240000, 12800000};  // halves
  const size_t objOff[3]    = {0, 120000, 150000};      // floats
  const size_t deltasOff[3] = {0, 480000, 600000};      // floats

  char* ws = (char*)d_ws;
  size_t off = 0;
  auto take = [&](size_t bytes) { size_t o = off; off = (off + bytes + 255) & ~(size_t)255; return o; };
  _Float16* featH = (_Float16*)(ws + take(13440000ull * 2));
  _Float16* tH    = (_Float16*)(ws + take(13440000ull * 2));
  _Float16* wpt   = (_Float16*)(ws + take(589824ull * 2));
  float* objBuf   = (float*)(ws + take(157500ull * 4));
  float* deltaBuf = (float*)(ws + take(630000ull * 4));
  float* lvlBoxes = (float*)(ws + take(12ull * TK * 4 * 4));
  float* lvlScore = (float*)(ws + take(12ull * TK * 4));
  float* zpage    = (float*)(ws + take(256));

  // zero page for halo reads
  zero_page<<<1, 64, 0, stream>>>(zpage);

  // features -> fp16 NHWC
  for (int l = 0; l < 3; ++l) {
    int total = 4 * HWs[l] * 256;
    cvt_feat<<<(total + 255) / 256, 256, 0, stream>>>(featsF[l], featH + featHOff[l], HWs[l], total);
  }
  // pack weights
  pack_w<<<(589824 + 255) / 256, 256, 0, stream>>>(conv_w, wpt, 589824);

  // conv (WMMA, async-LDS pipeline) per level
  for (int l = 0; l < 3; ++l) {
    int M = 4 * HWs[l];
    dim3 grid((M + BM - 1) / BM, 256 / BN);
    conv3x3_wmma<<<grid, 256, 0, stream>>>(featH + featHOff[l], wpt, conv_b,
                                           (const _Float16*)zpage,
                                           tH + featHOff[l], M, HWs[l], Hs[l], Ws[l]);
  }

  // heads per level
  for (int l = 0; l < 3; ++l) {
    int M = 4 * HWs[l];
    heads<<<(M + 7) / 8, 256, 0, stream>>>(tH + featHOff[l], cls_w, cls_b, bbox_w, bbox_b,
                                           objBuf + objOff[l], deltaBuf + deltasOff[l],
                                           M, HWs[l]);
  }

  // per-level top-k + decode (4 images per launch)
  for (int l = 0; l < 3; ++l) {
    int L = ANUM * HWs[l];
    topk_level<<<4, 1024, 0, stream>>>(objBuf + objOff[l], deltaBuf + deltasOff[l],
                                       anchors[l], L,
                                       lvlBoxes + (size_t)l * 4 * TK * 4,
                                       lvlScore + (size_t)l * 4 * TK);
  }

  // NMS over all 12 (level,image) slots
  nms_kernel<<<12, 1024, 0, stream>>>(lvlBoxes, lvlScore);

  // final per-image top-1000
  final_topk<<<4, 1024, 0, stream>>>(lvlBoxes, lvlScore, (float*)d_out);
}
